// Label_Encoder_23407571763537
// MI455X (gfx1250) — matrix operand order repaired and checked by reading, not compiled, not run on hardware
//
#include <hip/hip_runtime.h>
#include <cstdint>
#include <cstddef>

// ---------------------------------------------------------------------------
// Model dims
// ---------------------------------------------------------------------------
#define B_  16
#define L_  1024
#define E_  128
#define DI_ 256
#define DS_ 48
#define K_  4
#define DTR_ 8
#define XD_ 104            // DTR + 2*DS
#define XDP_ 112           // padded to multiple of 16
#define BL_ (B_ * L_)
#define HALO_ 32           // max conv reach: 2*dil_max = 32
#define LP_ (L_ + 2 * HALO_)   // padded time extent = 1088

// ---------------------------------------------------------------------------
// bf16 helpers / WMMA fragment types
// ---------------------------------------------------------------------------
typedef __attribute__((ext_vector_type(16))) __bf16 v16bf;
typedef __attribute__((ext_vector_type(8)))  float  v8f;

struct alignas(16) US8 { unsigned short u[8]; };

union FragBF {
    v16bf          v;
    US8            h[2];
    unsigned short u16[16];
};

static __device__ __forceinline__ unsigned short f2bf(float f) {
    union { float f; uint32_t u; } cv; cv.f = f;
    uint32_t r = (cv.u + 0x7FFFu + ((cv.u >> 16) & 1u)) >> 16;
    return (unsigned short)r;
}

static __device__ __forceinline__ float silu(float v) {
    return v / (1.f + __expf(-v));
}

// ---------------------------------------------------------------------------
// Front-end: conv0 (1 -> 128 channels, k=5, pad 2).  h layout (B,L,E) channels-last.
// Branch-free: clamped address + value select (v_cndmask, no exec branches).
// ---------------------------------------------------------------------------
__global__ void k_conv0(const float* __restrict__ x, const float* __restrict__ w0,
                        const float* __restrict__ b0, float* __restrict__ h) {
    int i = blockIdx.x * blockDim.x + threadIdx.x;
    if (i >= B_ * L_ * E_) return;
    int e = i & (E_ - 1);
    int l = (i >> 7) & (L_ - 1);
    int b = i >> 17;
    float s = b0[e];
#pragma unroll
    for (int t = 0; t < 5; ++t) {
        int ls = l + t - 2;
        int lc = ls < 0 ? 0 : (ls > L_ - 1 ? L_ - 1 : ls);
        float v = x[b * L_ + lc];
        s += (ls >= 0 && ls < L_) ? w0[e * 5 + t] * v : 0.f;
    }
    h[i] = s;
}

// gelu(exact) -> bf16 into time-padded channels-last layout (B,LP_,E), halo zero-filled.
__global__ void k_gelu_bf16_pad(const float* __restrict__ h, unsigned short* __restrict__ o) {
    int i = blockIdx.x * blockDim.x + threadIdx.x;
    if (i >= B_ * LP_ * E_) return;
    int e  = i & (E_ - 1);
    int bl = i >> 7;              // b*LP_ + lp
    int lp = bl % LP_;
    int b  = bl / LP_;
    int l  = lp - HALO_;
    unsigned short r = 0;
    if (l >= 0 && l < L_) {
        float v = h[((size_t)b * L_ + l) * E_ + e];
        r = f2bf(0.5f * v * (1.f + erff(v * 0.70710678118f)));
    }
    o[i] = r;
}

// conv_w (E,E,5) [e][c][t] fp32  ->  wA (5,E,E) [t][e][c] bf16
__global__ void k_pack_convw(const float* __restrict__ w, unsigned short* __restrict__ dst) {
    int i = blockIdx.x * blockDim.x + threadIdx.x;
    if (i >= 5 * E_ * E_) return;
    int c = i & (E_ - 1);
    int e = (i >> 7) & (E_ - 1);
    int t = i / (E_ * E_);
    dst[i] = f2bf(w[(e * E_ + c) * 5 + t]);
}

// Implicit-GEMM dilated conv via WMMA bf16, channels-last:
// out[b,l,e] = sum_{c,t} w[e,c,t] * in_pad[b, HALO_+l+(t-2)*dil, c]
// D tile: M = l (16), N = e (16). A = activations (contiguous in c),
// B = packed weights [t][e][c] (contiguous in c). All loads are 16B vector loads.
__global__ __launch_bounds__(32) void k_dilconv(
    const unsigned short* __restrict__ hp,  // (B,LP_,E) bf16, gelu applied, halo=0
    const unsigned short* __restrict__ wA,  // (5,E,E) bf16 [t][e][c]
    const float* __restrict__ bias,         // (E)
    float* __restrict__ out,                // (B,L,E)
    int dil) {
    int lane = threadIdx.x;
    int half = lane >> 4;
    int lm   = lane & 15;
    int l0 = blockIdx.x * 16, e0 = blockIdx.y * 16, b = blockIdx.z;
    v8f acc = {};
    const unsigned short* hpb = hp + (size_t)b * LP_ * E_;
#pragma unroll
    for (int t = 0; t < 5; ++t) {
        // A row for this lane: time index (always inside padded buffer)
        int lrow = HALO_ + l0 + lm + (t - 2) * dil;
        const unsigned short* arow = hpb + (size_t)lrow * E_;
        const unsigned short* wrow = wA + ((size_t)t * E_ + (e0 + lm)) * E_;
        for (int c0 = 0; c0 < E_; c0 += 32) {
            FragBF a, bb;
            // A (16x32, M=l, K=c): lane<16 holds K {0..7,16..23}, lane>=16 {8..15,24..31}
            int ka = c0 + half * 8;
            a.h[0] = *(const US8*)(arow + ka);
            a.h[1] = *(const US8*)(arow + ka + 16);
            // B (32x16, K=c, N=e): lane<16 holds K {0..7,16..23}, lane>=16 {8..15,24..31}; N = lane&15
            int kb = c0 + half * 8;
            bb.h[0] = *(const US8*)(wrow + kb);
            bb.h[1] = *(const US8*)(wrow + kb + 16);
            acc = __builtin_amdgcn_wmma_f32_16x16x32_bf16(false, a.v, false, bb.v,
                                                          (short)0, acc, false, false);
        }
    }
    float bv = bias[e0 + lm];
#pragma unroll
    for (int v = 0; v < 8; ++v) {
        int l = l0 + v + half * 8;
        out[((size_t)b * L_ + l) * E_ + e0 + lm] = acc[v] + bv;
    }
}

// residual (broadcast x over channels) + pos_emb; writes residual stream (d_out)
__global__ void k_resid_pos(const float* __restrict__ h, const float* __restrict__ x,
                            const float* __restrict__ pos, float* __restrict__ o) {
    int i = blockIdx.x * blockDim.x + threadIdx.x;
    if (i >= B_ * L_ * E_) return;
    int e = i & (E_ - 1);
    int l = (i >> 7) & (L_ - 1);
    int b = i >> 17;
    o[i] = h[i] + x[b * L_ + l] + pos[l * E_ + e];
}

// ---------------------------------------------------------------------------
// LayerNorm over E=128, output bf16 (feeds in_proj GEMM)
// ---------------------------------------------------------------------------
__global__ void k_ln_bf16(const float* __restrict__ x, const float* __restrict__ w,
                          const float* __restrict__ b, unsigned short* __restrict__ o,
                          int rows) {
    int r = blockIdx.x * blockDim.x + threadIdx.x;
    if (r >= rows) return;
    const float* xr = x + (size_t)r * E_;
    float mu = 0.f;
    for (int e = 0; e < E_; ++e) mu += xr[e];
    mu *= (1.f / E_);
    float var = 0.f;
    for (int e = 0; e < E_; ++e) { float d = xr[e] - mu; var += d * d; }
    var *= (1.f / E_);
    float rs = rsqrtf(var + 1e-5f);
    unsigned short* orow = o + (size_t)r * E_;
    for (int e = 0; e < E_; ++e)
        orow[e] = f2bf((xr[e] - mu) * rs * w[e] + b[e]);
}

// generic fp32 [N][K] -> bf16 [Npad][K] (zero pad rows N..Npad-1)
__global__ void k_pack_w(const float* __restrict__ src, unsigned short* __restrict__ dst,
                         int N, int K, int Npad) {
    int i = blockIdx.x * blockDim.x + threadIdx.x;
    if (i >= Npad * K) return;
    int n = i / K;
    dst[i] = (n < N) ? f2bf(src[i]) : (unsigned short)0;
}

// ---------------------------------------------------------------------------
// Generic bf16 WMMA GEMM: C[M,N] = A[M,K] * W[N,K]^T (+bias[n]) (+=C if accum)
// A,W bf16 row-major; C fp32 row-major with leading dim ldc. One wave / 16x16 tile.
// ---------------------------------------------------------------------------
__global__ __launch_bounds__(32) void k_gemm_bf16(
    const unsigned short* __restrict__ A, const unsigned short* __restrict__ W,
    float* __restrict__ C, const float* __restrict__ bias,
    int K, int ldc, int accum) {
    int lane = threadIdx.x;
    int half = lane >> 4;
    int lm   = lane & 15;
    int n0 = blockIdx.x * 16, m0 = blockIdx.y * 16;
    v8f acc = {};
    const unsigned short* arow = A + (size_t)(m0 + lm) * K;
    const unsigned short* wrow = W + (size_t)(n0 + lm) * K;
    for (int k0 = 0; k0 < K; k0 += 32) {
        FragBF a, b;
        int ka = k0 + half * 8;
        a.h[0] = *(const US8*)(arow + ka);
        a.h[1] = *(const US8*)(arow + ka + 16);
        int kb = k0 + half * 8;
        b.h[0] = *(const US8*)(wrow + kb);
        b.h[1] = *(const US8*)(wrow + kb + 16);
        acc = __builtin_amdgcn_wmma_f32_16x16x32_bf16(false, a.v, false, b.v,
                                                      (short)0, acc, false, false);
    }
    int n = n0 + lm;
    float bv = bias ? bias[n] : 0.f;
#pragma unroll
    for (int v = 0; v < 8; ++v) {
        int m = m0 + v + half * 8;
        size_t idx = (size_t)m * ldc + n;
        float r = acc[v] + bv;
        if (accum) r += C[idx];
        C[idx] = r;
    }
}

// ---------------------------------------------------------------------------
// Mamba direction helpers
// ---------------------------------------------------------------------------
// flip xpart in time: xrev[b,l,d] = xz[b, L-1-l, d]  (xz rows are width 512)
__global__ void k_flip_xpart(const float* __restrict__ xz, float* __restrict__ xrev) {
    int i = blockIdx.x * blockDim.x + threadIdx.x;
    if (i >= B_ * L_ * DI_) return;
    int d = i & (DI_ - 1);
    int l = (i >> 8) & (L_ - 1);
    int b = i >> 18;
    xrev[i] = xz[((size_t)(b * L_ + (L_ - 1 - l))) * (2 * DI_) + d];
}

// depthwise causal conv (k=4) + silu; input row stride rs (512 for fwd view into xz)
// branch-free boundary via clamped address + value select
__global__ void k_dwconv_silu(const float* __restrict__ in, int rs,
                              const float* __restrict__ cw, const float* __restrict__ cb,
                              float* __restrict__ xc) {
    int i = blockIdx.x * blockDim.x + threadIdx.x;
    if (i >= B_ * L_ * DI_) return;
    int d = i & (DI_ - 1);
    int l = (i >> 8) & (L_ - 1);
    int b = i >> 18;
    float s = cb[d];
#pragma unroll
    for (int t = 0; t < K_; ++t) {
        int ls = l - (K_ - 1) + t;
        int lc = ls < 0 ? 0 : ls;
        float v = in[((size_t)(b * L_ + lc)) * rs + d];
        s += (ls >= 0) ? cw[d * K_ + t] * v : 0.f;
    }
    xc[i] = silu(s);
}

__global__ void k_f2bf(const float* __restrict__ a, unsigned short* __restrict__ o, int n) {
    int i = blockIdx.x * blockDim.x + threadIdx.x;
    if (i >= n) return;
    o[i] = f2bf(a[i]);
}

// ---------------------------------------------------------------------------
// Selective scan: one block per batch element, 256 threads = DI channels.
// Per timestep the 112-float (dt|B|C) row is staged through LDS and shared by
// all channels; state h[48] and A[48] live in VGPRs. Fuses softplus(delta).
// ---------------------------------------------------------------------------
__global__ __launch_bounds__(256) void k_scan(
    const float* __restrict__ xdbl,   // (B,L,112)
    const float* __restrict__ xc,     // (B,L,256)
    const float* __restrict__ dtw,    // (256,8)
    const float* __restrict__ dtb,    // (256)
    const float* __restrict__ Alog,   // (256,48)
    const float* __restrict__ Dp,     // (256)
    float* __restrict__ y)            // (B,L,256)
{
    __shared__ float sm[XDP_];
    int b = blockIdx.x;
    int d = threadIdx.x;

    float Arow[DS_], h[DS_], w8[DTR_];
#pragma unroll
    for (int s = 0; s < DS_; ++s) { Arow[s] = -__expf(Alog[d * DS_ + s]); h[s] = 0.f; }
#pragma unroll
    for (int j = 0; j < DTR_; ++j) w8[j] = dtw[d * DTR_ + j];
    float db = dtb[d], Dd = Dp[d];

    const float* xcb = xc + (size_t)b * L_ * DI_;
    float*       yb  = y  + (size_t)b * L_ * DI_;

    for (int l = 0; l < L_; ++l) {
        const float* row = xdbl + ((size_t)b * L_ + l) * XDP_;
        if (d < XDP_) sm[d] = row[d];
        __syncthreads();
        if (l + 1 < L_ && d < XDP_) __builtin_prefetch(row + XDP_ + d, 0, 1);

        float t = db;
#pragma unroll
        for (int j = 0; j < DTR_; ++j) t += sm[j] * w8[j];
        float delta = (t > 20.f) ? t : log1pf(__expf(t));   // softplus
        float xv = xcb[l * DI_ + d];
        float sc = delta * xv;
        float acc = 0.f;
#pragma unroll
        for (int s = 0; s < DS_; ++s) {
            h[s] = __expf(delta * Arow[s]) * h[s] + sc * sm[DTR_ + s];
            acc += h[s] * sm[DTR_ + DS_ + s];
        }
        yb[l * DI_ + d] = acc + xv * Dd;
        __syncthreads();
    }
}

// (yf + flip(yr)) * silu(z) -> bf16 (feeds out_proj GEMM)
__global__ void k_combine(const float* __restrict__ yf, const float* __restrict__ yr,
                          const float* __restrict__ xz, unsigned short* __restrict__ o) {
    int i = blockIdx.x * blockDim.x + threadIdx.x;
    if (i >= B_ * L_ * DI_) return;
    int d = i & (DI_ - 1);
    int l = (i >> 8) & (L_ - 1);
    int b = i >> 18;
    float z = xz[((size_t)(b * L_ + l)) * (2 * DI_) + DI_ + d];
    float v = (yf[i] + yr[((size_t)(b * L_ + (L_ - 1 - l))) * DI_ + d]) * silu(z);
    o[i] = f2bf(v);
}

// ---------------------------------------------------------------------------
// Host driver
// ---------------------------------------------------------------------------
extern "C" void kernel_launch(void* const* d_in, const int* in_sizes, int n_in,
                              void* d_out, int out_size, void* d_ws, size_t ws_size,
                              hipStream_t stream) {
    (void)in_sizes; (void)n_in; (void)out_size; (void)ws_size;

    const float* x        = (const float*)d_in[0];
    const float* conv_w0  = (const float*)d_in[1];
    const float* conv_b0  = (const float*)d_in[2];
    const float* conv_w   = (const float*)d_in[3];   // (4,E,E,5)
    const float* conv_b   = (const float*)d_in[4];   // (4,E)
    const float* pos_emb  = (const float*)d_in[5];
    const float* ln_w     = (const float*)d_in[6];
    const float* ln_b     = (const float*)d_in[7];
    const float* in_proj  = (const float*)d_in[8];   // (2,512,128)
    const float* out_proj = (const float*)d_in[9];   // (2,128,256)
    const float* cw_f   = (const float*)d_in[10];
    const float* cb_f   = (const float*)d_in[11];
    const float* xp_f   = (const float*)d_in[12];
    const float* dtw_f  = (const float*)d_in[13];
    const float* dtb_f  = (const float*)d_in[14];
    const float* Alog_f = (const float*)d_in[15];
    const float* D_f    = (const float*)d_in[16];
    const float* cw_r   = (const float*)d_in[17];
    const float* cb_r   = (const float*)d_in[18];
    const float* xp_r   = (const float*)d_in[19];
    const float* dtw_r  = (const float*)d_in[20];
    const float* dtb_r  = (const float*)d_in[21];
    const float* Alog_r = (const float*)d_in[22];
    const float* D_r    = (const float*)d_in[23];

    float* resid = (float*)d_out;  // (B,L,E) residual stream, final output

    // ---- workspace carve ----
    char* base = (char*)d_ws;
    size_t off = 0;
    auto alloc = [&](size_t bytes) -> void* {
        void* r = base + off;
        off += (bytes + 255) & ~(size_t)255;
        return r;
    };
    float*          ha    = (float*)alloc((size_t)B_ * L_ * E_ * 4);
    float*          hb    = (float*)alloc((size_t)B_ * L_ * E_ * 4);
    unsigned short* hp16  = (unsigned short*)alloc((size_t)B_ * LP_ * E_ * 2);  // padded, channels-last
    unsigned short* wbuf  = (unsigned short*)alloc((size_t)5 * E_ * E_ * 2);    // reused for all packed weights
    unsigned short* ubf   = (unsigned short*)alloc((size_t)BL_ * E_ * 2);
    float*          xz    = (float*)alloc((size_t)BL_ * 2 * DI_ * 4);
    float*          xrev  = (float*)alloc((size_t)BL_ * DI_ * 4);
    float*          xc    = (float*)alloc((size_t)BL_ * DI_ * 4);
    unsigned short* xcb16 = (unsigned short*)alloc((size_t)BL_ * DI_ * 2);      // also holds gated y in bf16
    float*          xdbl  = (float*)alloc((size_t)BL_ * XDP_ * 4);
    float*          yf    = (float*)alloc((size_t)BL_ * DI_ * 4);
    float*          yr    = (float*)alloc((size_t)BL_ * DI_ * 4);

    const int nBEL  = B_ * L_ * E_;
    const int nBELp = B_ * LP_ * E_;
    const int nBLD  = B_ * L_ * DI_;
    dim3 blk(256);

    // ---- front-end conv stack (channels-last) ----
    k_conv0<<<(nBEL + 255) / 256, blk, 0, stream>>>(x, conv_w0, conv_b0, ha);
    const int dils[4] = {2, 4, 8, 16};
    float* hcur = ha;
    float* hnxt = hb;
    for (int i = 0; i < 4; ++i) {
        k_gelu_bf16_pad<<<(nBELp + 255) / 256, blk, 0, stream>>>(hcur, hp16);
        k_pack_convw<<<(5 * E_ * E_ + 255) / 256, blk, 0, stream>>>(conv_w + (size_t)i * E_ * E_ * 5, wbuf);
        k_dilconv<<<dim3(L_ / 16, E_ / 16, B_), dim3(32), 0, stream>>>(hp16, wbuf, conv_b + i * E_, hnxt, dils[i]);
        float* t = hcur; hcur = hnxt; hnxt = t;
    }
    k_resid_pos<<<(nBEL + 255) / 256, blk, 0, stream>>>(hcur, x, pos_emb, resid);

    // ---- bimamba blocks ----
    for (int i = 0; i < 2; ++i) {
        const float* lw  = ln_w + i * E_;
        const float* lb  = ln_b + i * E_;
        const float* ipw = in_proj + (size_t)i * 2 * DI_ * E_;
        const float* opw = out_proj + (size_t)i * E_ * DI_;

        // layernorm -> bf16
        k_ln_bf16<<<(BL_ + 255) / 256, blk, 0, stream>>>(resid, lw, lb, ubf, BL_);
        // in_proj GEMM: (BL,128) @ (512,128)^T -> xz (BL,512)
        k_pack_w<<<(2 * DI_ * E_ + 255) / 256, blk, 0, stream>>>(ipw, wbuf, 2 * DI_, E_, 2 * DI_);
        k_gemm_bf16<<<dim3(2 * DI_ / 16, BL_ / 16), dim3(32), 0, stream>>>(
            ubf, wbuf, xz, nullptr, E_, 2 * DI_, 0);

        for (int dir = 0; dir < 2; ++dir) {
            const float* cw   = (dir == 0 ? cw_f   : cw_r)   + (size_t)i * DI_ * K_;
            const float* cb   = (dir == 0 ? cb_f   : cb_r)   + (size_t)i * DI_;
            const float* xp   = (dir == 0 ? xp_f   : xp_r)   + (size_t)i * XD_ * DI_;
            const float* dtw  = (dir == 0 ? dtw_f  : dtw_r)  + (size_t)i * DI_ * DTR_;
            const float* dtb  = (dir == 0 ? dtb_f  : dtb_r)  + (size_t)i * DI_;
            const float* Alog = (dir == 0 ? Alog_f : Alog_r) + (size_t)i * DI_ * DS_;
            const float* Dp   = (dir == 0 ? D_f    : D_r)    + (size_t)i * DI_;
            float* ydst = (dir == 0) ? yf : yr;

            const float* dwin = xz;
            int rs = 2 * DI_;
            if (dir == 1) {
                k_flip_xpart<<<(nBLD + 255) / 256, blk, 0, stream>>>(xz, xrev);
                dwin = xrev;
                rs = DI_;
            }
            // depthwise causal conv + silu
            k_dwconv_silu<<<(nBLD + 255) / 256, blk, 0, stream>>>(dwin, rs, cw, cb, xc);
            // x_dbl GEMM: (BL,256) @ (104->112,256)^T -> xdbl (BL,112)
            k_f2bf<<<(nBLD + 255) / 256, blk, 0, stream>>>(xc, xcb16, nBLD);
            k_pack_w<<<(XDP_ * DI_ + 255) / 256, blk, 0, stream>>>(xp, wbuf, XD_, DI_, XDP_);
            k_gemm_bf16<<<dim3(XDP_ / 16, BL_ / 16), dim3(32), 0, stream>>>(
                xcb16, wbuf, xdbl, nullptr, DI_, XDP_, 0);
            // selective scan (fused softplus(dt@dtw^T+dtb))
            k_scan<<<B_, dim3(DI_), 0, stream>>>(xdbl, xc, dtw, dtb, Alog, Dp, ydst);
        }

        // gate + combine directions -> bf16, then out_proj GEMM accumulated into residual
        k_combine<<<(nBLD + 255) / 256, blk, 0, stream>>>(yf, yr, xz, xcb16);
        k_pack_w<<<(E_ * DI_ + 255) / 256, blk, 0, stream>>>(opw, wbuf, E_, DI_, E_);
        k_gemm_bf16<<<dim3(E_ / 16, BL_ / 16), dim3(32), 0, stream>>>(
            xcb16, wbuf, resid, nullptr, DI_, E_, 1);
    }
}
